// QLlamaSdpaAttention_29317446762579
// MI455X (gfx1250) — 16-bit and 8-bit matrix operand order repaired and checked by reading, compiled, not run on hardware
//
#include <hip/hip_runtime.h>


typedef __attribute__((ext_vector_type(16))) _Float16 v16h;
typedef __attribute__((ext_vector_type(8)))  float    v8f;
typedef __attribute__((ext_vector_type(8)))  int      v8i;
typedef __attribute__((ext_vector_type(2)))  _Float16 h2;
typedef __attribute__((ext_vector_type(4)))  int      v4i;

#define GLOBAL_AS __attribute__((address_space(1)))
#define LDS_AS    __attribute__((address_space(3)))

#if defined(__has_builtin)
# if __has_builtin(__builtin_amdgcn_global_load_async_to_lds_b128)
#  define HAS_ASYNC_LDS 1
# endif
#endif
#ifndef HAS_ASYNC_LDS
# define HAS_ASYNC_LDS 0
#endif

union FragH {
    v16h v;
    unsigned int u[8];
};
union FragI {
    v8i v;
    unsigned int u[8];
};

static __device__ inline v8f vzero8() {
    v8f r;
#pragma unroll
    for (int i = 0; i < 8; ++i) r[i] = 0.0f;
    return r;
}
static __device__ inline v8i izero8() {
    v8i r;
#pragma unroll
    for (int i = 0; i < 8; ++i) r[i] = 0;
    return r;
}

static __device__ inline unsigned int pack2(float lo, float hi) {
    h2 p;
    p[0] = (_Float16)lo;
    p[1] = (_Float16)hi;
    return __builtin_bit_cast(unsigned int, p);
}

static __device__ inline v8f wmma_f16(const v16h& a, const v16h& b, const v8f& c) {
    return __builtin_amdgcn_wmma_f32_16x16x32_f16(
        false, a, false, b, (short)0, c, false, false);
}
static __device__ inline v8i wmma_iu8(const v8i& a, const v8i& b, const v8i& c) {
    return __builtin_amdgcn_wmma_i32_16x16x64_iu8(true, a, true, b, c, false, false);
}

static __device__ inline void cp16_g2l(const void* g, void* l) {
#if HAS_ASYNC_LDS
    void* gp = const_cast<void*>(g);
    __builtin_amdgcn_global_load_async_to_lds_b128(
        (GLOBAL_AS v4i*)gp, (LDS_AS v4i*)l, 0, 0);
#else
    *(float4*)l = *(const float4*)g;
#endif
}

template <int N>
static __device__ inline void wait_async() {
#if HAS_ASYNC_LDS
# if defined(__has_builtin) && __has_builtin(__builtin_amdgcn_s_wait_asynccnt)
    __builtin_amdgcn_s_wait_asynccnt(N);
# else
    asm volatile("s_wait_asynccnt %0" :: "n"(N));
# endif
#endif
}

static __device__ inline float halfmax16(float x) {
#if defined(__has_builtin) && __has_builtin(__builtin_amdgcn_ds_swizzle)
    x = fmaxf(x, __int_as_float(__builtin_amdgcn_ds_swizzle(__float_as_int(x), (8 << 10) | 0x1f)));
    x = fmaxf(x, __int_as_float(__builtin_amdgcn_ds_swizzle(__float_as_int(x), (4 << 10) | 0x1f)));
    x = fmaxf(x, __int_as_float(__builtin_amdgcn_ds_swizzle(__float_as_int(x), (2 << 10) | 0x1f)));
    x = fmaxf(x, __int_as_float(__builtin_amdgcn_ds_swizzle(__float_as_int(x), (1 << 10) | 0x1f)));
#else
#pragma unroll
    for (int off = 8; off >= 1; off >>= 1)
        x = fmaxf(x, __shfl_xor(x, off, 32));
#endif
    return x;
}

#define SEQ     2048
#define HID     4096
#define NH      32
#define NKV     8
#define HD      128
#define QKVN    6144
#define QMAXF   127.0f

static __device__ inline int kb8A(int g, int h) {
    return (g >> 1) * 16 + (g & 1) * 4 + 8 * h;
}
static __device__ inline int kb8B(int g, int h) {
    return (g < 4) ? (4 * g + 16 * h) : (32 + 4 * (g - 4) + 16 * h);
}

__global__ void zero_scales_kernel(int* sb) {
    if (threadIdx.x < 8) sb[threadIdx.x] = 0;
}

template <int MODE>
__global__ __launch_bounds__(256)
void gemm_wmma_kernel(const float* __restrict__ A32,
                      const _Float16* __restrict__ A16,
                      const float* __restrict__ Bq,
                      const float* __restrict__ Bk,
                      const float* __restrict__ Bv,
                      void* __restrict__ Cout,
                      int M, int N, int K) {
    __shared__ float    BsF[2][32][132];
    __shared__ float    AsF[2][64][36];
    __shared__ _Float16 AsH[2][64][40];

    const int tid  = threadIdx.x;
    const int wave = tid >> 5;
    const int lane = tid & 31;
    const int lan  = lane & 15;
    const int half = lane >> 4;

    const int m0 = blockIdx.y * 64;
    const int n0 = blockIdx.x * 128;
    const int m_off = (wave >> 1) * 16;
    const int n_off = (wave & 1) * 64;

    const float* Bsrc;
    int bstride, bcol0;
    if (MODE == 0) {
        if (n0 < 4096)      { Bsrc = Bq; bstride = 4096; bcol0 = n0; }
        else if (n0 < 5120) { Bsrc = Bk; bstride = 1024; bcol0 = n0 - 4096; }
        else                { Bsrc = Bv; bstride = 1024; bcol0 = n0 - 5120; }
    } else {
        Bsrc = Bq; bstride = N; bcol0 = n0;
    }

    auto stage = [&](int k0, int buf) {
        if (MODE == 0) {
#pragma unroll
            for (int i = 0; i < 2; ++i) {
                int s = tid + 256 * i;
                int r = s >> 3, c4 = s & 7;
                cp16_g2l(&A32[(size_t)(m0 + r) * K + k0 + c4 * 4],
                         &AsF[buf][r][c4 * 4]);
            }
        } else {
            int r = tid >> 2, c8 = tid & 3;
            cp16_g2l(&A16[(size_t)(m0 + r) * K + k0 + c8 * 8],
                     &AsH[buf][r][c8 * 8]);
        }
#pragma unroll
        for (int i = 0; i < 4; ++i) {
            int s = tid + 256 * i;
            int kk = s >> 5, c4 = s & 31;
            cp16_g2l(&Bsrc[(size_t)(k0 + kk) * bstride + bcol0 + c4 * 4],
                     &BsF[buf][kk][c4 * 4]);
        }
    };
    constexpr int NASYNC = (MODE == 0) ? 6 : 5;

    v8f acc[4];
#pragma unroll
    for (int t = 0; t < 4; ++t) acc[t] = vzero8();

    const int nk = K / 32;
    stage(0, 0);
    for (int it = 0; it < nk; ++it) {
        const int buf = it & 1;
        const int knext = (it + 1 < nk) ? (it + 1) * 32 : 0;
        stage(knext, buf ^ 1);
        wait_async<NASYNC>();
        __syncthreads();

        FragH a;
        const int m = m_off + lan;
#pragma unroll
        for (int g = 0; g < 8; ++g) {
            int kb = (g < 4) ? (2 * g + 8 * half) : (16 + 2 * (g - 4) + 8 * half);
            if (MODE == 0) {
                float2 f2 = *(const float2*)&AsF[buf][m][kb];
                a.u[g] = pack2(f2.x, f2.y);
            } else {
                a.u[g] = *(const unsigned int*)&AsH[buf][m][kb];
            }
        }
#pragma unroll
        for (int nt = 0; nt < 4; ++nt) {
            FragH b;
            const int n = n_off + nt * 16 + lan;
#pragma unroll
            for (int g = 0; g < 8; ++g) {
                int kk = (g < 4) ? (2 * g + 8 * half) : (16 + 2 * (g - 4) + 8 * half);
                b.u[g] = pack2(BsF[buf][kk][n], BsF[buf][kk + 1][n]);
            }
            acc[nt] = wmma_f16(a.v, b.v, acc[nt]);
        }
        __syncthreads();
    }

#pragma unroll
    for (int nt = 0; nt < 4; ++nt) {
#pragma unroll
        for (int v = 0; v < 8; ++v) {
            int row = m0 + m_off + v + 8 * half;
            int col = n0 + n_off + nt * 16 + lan;
            if (MODE == 0) {
                ((_Float16*)Cout)[(size_t)row * N + col] = (_Float16)acc[nt][v];
            } else {
                ((float*)Cout)[(size_t)row * N + col] = acc[nt][v];
            }
        }
    }
}

__global__ __launch_bounds__(256)
void rope_absmax_kernel(const _Float16* __restrict__ raw,
                        _Float16* __restrict__ outr,
                        int* __restrict__ scaleBits) {
    __shared__ int smax[3];
    if (threadIdx.x < 3) smax[threadIdx.x] = 0;
    __syncthreads();

    const float LOG1E4_O64 = 0.14391156831212787f;
    size_t base = ((size_t)blockIdx.x * 256 + threadIdx.x) * 8;
    float lm0 = 0.f, lm1 = 0.f, lm2 = 0.f;

#pragma unroll
    for (int i = 0; i < 8; ++i) {
        size_t idx = base + i;
        int p = (int)(idx / QKVN);
        int j = (int)(idx % QKVN);
        float x = (float)raw[idx];
        float val;
        if (j < 5120) {
            int d  = j & 127;
            int dm = d & 63;
            float invf = __expf(-(float)dm * LOG1E4_O64);
            float ang  = (float)p * invf;
            float c = __cosf(ang), s = __sinf(ang);
            float xr = (float)raw[(size_t)p * QKVN + ((d < 64) ? j + 64 : j - 64)];
            val = (d < 64) ? (x * c - xr * s) : (x * c + xr * s);
            if (j < 4096) lm0 = fmaxf(lm0, fabsf(val));
            else          lm1 = fmaxf(lm1, fabsf(val));
        } else {
            val = x;
            lm2 = fmaxf(lm2, fabsf(val));
        }
        outr[idx] = (_Float16)val;
    }
    atomicMax(&smax[0], __float_as_int(lm0));
    atomicMax(&smax[1], __float_as_int(lm1));
    atomicMax(&smax[2], __float_as_int(lm2));
    __syncthreads();
    if (threadIdx.x < 3) atomicMax(&scaleBits[threadIdx.x], smax[threadIdx.x]);
}

__global__ __launch_bounds__(256)
void quant_kernel(const _Float16* __restrict__ qkvr,
                  signed char* __restrict__ qi8,
                  signed char* __restrict__ ki8,
                  _Float16* __restrict__ vt,
                  const int* __restrict__ scaleBits) {
    float mq = __int_as_float(scaleBits[0]);
    float mk = __int_as_float(scaleBits[1]);
    float mv = __int_as_float(scaleBits[2]);
    float isq = (mq > 0.f) ? QMAXF / mq : 1.0f;
    float isk = (mk > 0.f) ? QMAXF / mk : 1.0f;
    float isv = (mv > 0.f) ? QMAXF / mv : 1.0f;

    size_t base = ((size_t)blockIdx.x * 256 + threadIdx.x) * 8;
#pragma unroll
    for (int i = 0; i < 8; ++i) {
        size_t idx = base + i;
        int p = (int)(idx / QKVN);
        int j = (int)(idx % QKVN);
        float x = (float)qkvr[idx];
        if (j < 4096) {
            qi8[(size_t)p * 4096 + j] = (signed char)(int)rintf(x * isq);
        } else if (j < 5120) {
            ki8[(size_t)p * 1024 + (j - 4096)] = (signed char)(int)rintf(x * isk);
        } else {
            vt[(size_t)(j - 5120) * SEQ + p] = (_Float16)rintf(x * isv);
        }
    }
}

__global__ __launch_bounds__(128)
void attn_kernel(const signed char* __restrict__ qi8,
                 const signed char* __restrict__ ki8,
                 const _Float16* __restrict__ vt,
                 const int* __restrict__ scaleBits,
                 _Float16* __restrict__ attnh) {
    __shared__ _Float16 sP[4][16][32];

    const int tid  = threadIdx.x;
    const int wave = tid >> 5;
    const int lane = tid & 31;
    const int lan  = lane & 15;
    const int half = lane >> 4;

    const int unit  = blockIdx.x * 4 + wave;
    const int head  = unit >> 7;
    const int qt    = unit & 127;
    const int qbase = qt * 16;
    const int kvh   = head >> 2;

    float mq = __int_as_float(scaleBits[0]);
    float mk = __int_as_float(scaleBits[1]);
    float mv = __int_as_float(scaleBits[2]);
    float sq = (mq > 0.f) ? mq / QMAXF : 1.0f;
    float sk = (mk > 0.f) ? mk / QMAXF : 1.0f;
    float sv = (mv > 0.f) ? mv / QMAXF : 1.0f;
    const float sfac = sq * sk * 0.08838834764831845f;

    const unsigned int* q8U = (const unsigned int*)qi8;
    const unsigned int* k8U = (const unsigned int*)ki8;
    const unsigned int* vU  = (const unsigned int*)vt;

    FragH ones;
#pragma unroll
    for (int g = 0; g < 8; ++g) ones.u[g] = 0x3C003C00u;

    FragI aq[2];
    {
        const int row = qbase + lan;
#pragma unroll
        for (int kc = 0; kc < 2; ++kc) {
            const int colb = head * HD + kc * 64;
#pragma unroll
            for (int g = 0; g < 8; ++g)
                aq[kc].u[g] = q8U[((size_t)row * 4096 + colb + kb8A(g, half)) >> 2];
        }
    }

    v8f o[8];
#pragma unroll
    for (int nt = 0; nt < 8; ++nt) o[nt] = vzero8();
    v8f osum = vzero8();
    float mrow[8];
#pragma unroll
    for (int v = 0; v < 8; ++v) mrow[v] = -1e30f;

    const int nkc = (qbase + 15) / 32;
    for (int kb = 0; kb <= nkc; ++kb) {
        const int kbase = kb * 32;

        v8i s0i = izero8(), s1i = izero8();
#pragma unroll
        for (int kc = 0; kc < 2; ++kc) {
            const int colb = kvh * HD + kc * 64;
            FragI b0, b1;
            const int kr0 = kbase + lan;
#pragma unroll
            for (int g = 0; g < 8; ++g) {
                int kk = colb + kb8A(g, half);
                b0.u[g] = k8U[((size_t)kr0 * 1024 + kk) >> 2];
                b1.u[g] = k8U[((size_t)(kr0 + 16) * 1024 + kk) >> 2];
            }
            s0i = wmma_iu8(aq[kc].v, b0.v, s0i);
            s1i = wmma_iu8(aq[kc].v, b1.v, s1i);
        }

        float fvec[8], p0v[8], p1v[8];
#pragma unroll
        for (int v = 0; v < 8; ++v) {
            int row  = qbase + v + 8 * half;
            int key0 = kbase + lan;
            float x0 = (key0 > row)      ? -1e9f : (float)s0i[v] * sfac;
            float x1 = (key0 + 16 > row) ? -1e9f : (float)s1i[v] * sfac;

            float mx   = halfmax16(fmaxf(x0, x1));
            float newm = fmaxf(mrow[v], mx);
            float fac  = __expf(mrow[v] - newm);
            mrow[v] = newm;
            p0v[v] = __expf(x0 - newm);
            p1v[v] = __expf(x1 - newm);
            fvec[v] = fac;
        }
#pragma unroll
        for (int nt = 0; nt < 8; ++nt)
#pragma unroll
            for (int v = 0; v < 8; ++v) o[nt][v] *= fvec[v];
#pragma unroll
        for (int v = 0; v < 8; ++v) osum[v] *= fvec[v];

#pragma unroll
        for (int v = 0; v < 8; ++v) {
            int row = v + 8 * half;
            sP[wave][row][lan]      = (_Float16)p0v[v];
            sP[wave][row][16 + lan] = (_Float16)p1v[v];
        }
        __asm__ __volatile__("" ::: "memory");
        FragH ap;
#pragma unroll
        for (int g = 0; g < 8; ++g) {
            int kk = (g < 4) ? (2 * g + 8 * half) : (16 + 2 * (g - 4) + 8 * half);
            ap.u[g] = *(const unsigned int*)&sP[wave][lan][kk];
        }

        osum = wmma_f16(ap.v, ones.v, osum);

#pragma unroll
        for (int nt = 0; nt < 8; ++nt) {
            FragH bv;
            const int dim = nt * 16 + lan;
#pragma unroll
            for (int g = 0; g < 8; ++g) {
                int kk = (g < 4) ? (2 * g + 8 * half) : (16 + 2 * (g - 4) + 8 * half);
                bv.u[g] = vU[((size_t)(kvh * HD + dim) * SEQ + kbase + kk) >> 1];
            }
            o[nt] = wmma_f16(ap.v, bv.v, o[nt]);
        }
    }

    float inv[8];
#pragma unroll
    for (int v = 0; v < 8; ++v) inv[v] = sv / osum[v];
#pragma unroll
    for (int nt = 0; nt < 8; ++nt) {
#pragma unroll
        for (int v = 0; v < 8; ++v) {
            int row = qbase + v + 8 * half;
            int col = head * HD + nt * 16 + lan;
            attnh[(size_t)row * HID + col] = (_Float16)(o[nt][v] * inv[v]);
        }
    }
}

extern "C" void kernel_launch(void* const* d_in, const int* in_sizes, int n_in,
                              void* d_out, int out_size, void* d_ws, size_t ws_size,
                              hipStream_t stream) {
    (void)in_sizes; (void)n_in; (void)out_size; (void)ws_size;
    const float* hs = (const float*)d_in[0];
    const float* wq = (const float*)d_in[3];
    const float* wk = (const float*)d_in[4];
    const float* wv = (const float*)d_in[5];
    const float* wo = (const float*)d_in[6];
    float* out = (float*)d_out;

    char* w = (char*)d_ws;
    int*         scaleBits = (int*)w;
    _Float16*    qkvraw    = (_Float16*)(w + 256);
    _Float16*    qkvr      = qkvraw + (size_t)SEQ * QKVN;
    _Float16*    vt        = qkvr   + (size_t)SEQ * QKVN;
    signed char* qi8       = (signed char*)(vt + (size_t)NKV * HD * SEQ);
    signed char* ki8       = qi8 + (size_t)SEQ * 4096;
    _Float16*    attnh     = qkvraw;

    zero_scales_kernel<<<1, 32, 0, stream>>>(scaleBits);

    gemm_wmma_kernel<0><<<dim3(QKVN / 128, SEQ / 64), 256, 0, stream>>>(
        hs, nullptr, wq, wk, wv, (void*)qkvraw, SEQ, QKVN, HID);

    const int elems = SEQ * QKVN;
    rope_absmax_kernel<<<elems / (256 * 8), 256, 0, stream>>>(qkvraw, qkvr, scaleBits);

    quant_kernel<<<elems / (256 * 8), 256, 0, stream>>>(qkvr, qi8, ki8, vt, scaleBits);

    attn_kernel<<<(NH * (SEQ / 16)) / 4, 128, 0, stream>>>(qi8, ki8, vt, scaleBits, attnh);

    gemm_wmma_kernel<1><<<dim3(HID / 128, SEQ / 64), 256, 0, stream>>>(
        nullptr, attnh, wo, nullptr, nullptr, (void*)out, SEQ, HID, HID);
}
